// MultiHeadSelfAttention_84456236908902
// MI455X (gfx1250) — matrix operand order repaired and checked by reading, not compiled, not run on hardware
//
#include <hip/hip_runtime.h>

typedef __attribute__((ext_vector_type(16))) _Float16 v16h;
typedef __attribute__((ext_vector_type(8)))  _Float16 v8h;
typedef __attribute__((ext_vector_type(4)))  _Float16 v4h;
typedef __attribute__((ext_vector_type(2)))  _Float16 v2h;
typedef __attribute__((ext_vector_type(8)))  float    v8f;
typedef __attribute__((ext_vector_type(4)))  float    v4f;
typedef __attribute__((ext_vector_type(4)))  unsigned int v4u;
typedef __attribute__((ext_vector_type(8)))  int      v8i;
typedef __attribute__((ext_vector_type(4)))  int      v4i;

#define EMBED   1024
#define HEADS   16
#define HDIM    64
#define BATCH   4
#define SEQ     2048
#define MROWS   (BATCH*SEQ)   /* 8192 */

__device__ __forceinline__ v16h cat8(v8h lo, v8h hi) {
  v16h r;
#pragma unroll
  for (int i = 0; i < 8; ++i) { r[i] = lo[i]; r[i + 8] = hi[i]; }
  return r;
}

__device__ __forceinline__ v8f wmma_f16(v16h a, v16h b, v8f c) {
  // D = A(16x32 f16) * B(32x16 f16) + C(16x16 f32)
  return __builtin_amdgcn_wmma_f32_16x16x32_f16(false, a, false, b, (short)0, c, false, false);
}

// ---------------------------------------------------------------------------
// TDM: DMA a [64 x 64] f16 tile (row-major) from global into LDS, padding
// LDS rows to 72 halfs (pad 4 dwords every 32 dwords) to kill bank conflicts.
// D# bitfields per CDNA5 ISA ch.8. This toolchain exposes the 6-arg builtin:
//   (uint32x4 g0, int32x8 g1, int32x4 g2, int32x4 g3, int32x8 g4, i32 cpol)
// ---------------------------------------------------------------------------
#if __has_builtin(__builtin_amdgcn_tensor_load_to_lds)
#define HAVE_TDM 1
__device__ __forceinline__ void tdm_load_tile64x64_f16(
    unsigned lds_byte, const _Float16* gptr,
    unsigned tensor_d0, unsigned tensor_d1, unsigned stride0)
{
  unsigned long long ga = (unsigned long long)(size_t)gptr;
  v4u g0;
  g0[0] = 1u;                                        // count=1 valid user D#
  g0[1] = lds_byte;                                  // lds_addr
  g0[2] = (unsigned)(ga & 0xffffffffu);              // global_addr[31:0]
  g0[3] = (unsigned)((ga >> 32) & 0x01ffffffu)       // global_addr[56:32]
        | 0x80000000u;                               // type=2 ("image")
  v8i g1;
  g1[0] = (int)((1u << 16)                           // data_size=1 -> 2 bytes
        |       (1u << 20)                           // pad_enable
        |       (4u << 22)                           // pad_interval: 32 dwords
        |       (3u << 25));                         // pad_amount: 4 dwords
  g1[1] = (int)((tensor_d0 & 0xffffu) << 16);        // tensor_dim0[15:0]
  g1[2] = (int)(((tensor_d0 >> 16) & 0xffffu)
        |       ((tensor_d1 & 0xffffu) << 16));      // dim0[31:16] | dim1[15:0]
  g1[3] = (int)(((tensor_d1 >> 16) & 0xffffu)
        |       (64u << 16));                        // dim1[31:16] | tile_dim0=64
  g1[4] = (int)64u;                                  // tile_dim1=64, tile_dim2=0
  g1[5] = (int)stride0;                              // tensor_dim0_stride[31:0]
  g1[6] = 0;                                         // stride[47:32]=0, dim1_stride lo
  g1[7] = 0;
  v4i z4 = {0, 0, 0, 0};
  v8i z8 = {0, 0, 0, 0, 0, 0, 0, 0};
  __builtin_amdgcn_tensor_load_to_lds(g0, g1, z4, z4, z8, 0);
}
#else
#define HAVE_TDM 0
#endif

// ---------------------------------------------------------------------------
// Tiled WMMA GEMM:  C[M,N] = A[M,K] * B[K,N] + bias[N]
//   block tile 128x128, 256 threads (8 waves), wave tile 64x32 (4x2 WMMA accs)
// ---------------------------------------------------------------------------
template<bool A_IS_F32, bool OUT_IS_F16>
__global__ void __launch_bounds__(256)
gemm128_wmma(const void* __restrict__ Av, const float* __restrict__ Bw,
             const float* __restrict__ bias, void* __restrict__ Cv,
             int M, int N, int K)
{
  __shared__ _Float16 As[128][40];   // row-major, 80B rows (16B-aligned)
  __shared__ _Float16 Bs[128][40];   // TRANSPOSED: Bs[n][k]

  const int tid   = threadIdx.x;
  const int lane  = tid & 31;
  const int wid   = tid >> 5;            // 0..7
  const int waveM = wid >> 2;            // 0..1 (64 rows)
  const int waveN = wid & 3;             // 0..3 (32 cols)
  const int lr    = lane & 15;
  const int hf    = lane >> 4;

  const int m0 = blockIdx.y * 128;
  const int n0 = blockIdx.x * 128;

  v8f acc[4][2] = {};

  const int kTiles = K >> 5;             // K/32
  for (int kt = 0; kt < kTiles; ++kt) {
    // ---- L2 prefetch of next B tile (global_prefetch_b8) ----
    if (kt + 1 < kTiles) {
      __builtin_prefetch((const char*)Bw +
          ((size_t)((kt + 1) * 32 + (tid >> 4) * 2) * N + n0 + (tid & 15) * 8) * 4, 0, 1);
    }
    // ---- stage A tile (128 x 32): one b64 store per 4 elems ----
    {
      const int colb = (tid & 7) * 4;
#pragma unroll
      for (int p = 0; p < 4; ++p) {
        const int row = p * 32 + (tid >> 3);
        if (A_IS_F32) {
          const float* ap = (const float*)Av + (size_t)(m0 + row) * K + kt * 32 + colb;
          v4f v = *(const v4f*)ap;
          v4h hh;
#pragma unroll
          for (int e = 0; e < 4; ++e) hh[e] = (_Float16)v[e];
          *(v4h*)(&As[row][colb]) = hh;
        } else {
          const _Float16* ap = (const _Float16*)Av + (size_t)(m0 + row) * K + kt * 32 + colb;
          *(v4h*)(&As[row][colb]) = *(const v4h*)ap;
        }
      }
    }
    // ---- stage B tile (32 x 128) transposed, k-pairs -> packed b32 stores ----
    {
      const int kp = (tid >> 4) * 2;         // 0,2,..,30
      const int nb = (tid & 15) * 8;         // 0..120
      const float* bp = Bw + (size_t)(kt * 32 + kp) * N + n0 + nb;
      v4f r0a = *(const v4f*)bp;       v4f r0b = *(const v4f*)(bp + 4);
      v4f r1a = *(const v4f*)(bp + N); v4f r1b = *(const v4f*)(bp + N + 4);
#pragma unroll
      for (int e = 0; e < 8; ++e) {
        v2h p;
        p[0] = (_Float16)((e < 4) ? r0a[e] : r0b[e - 4]);
        p[1] = (_Float16)((e < 4) ? r1a[e] : r1b[e - 4]);
        *(v2h*)(&Bs[nb + e][kp]) = p;
      }
    }
    __syncthreads();

    // ---- fragments + WMMA ----
    v16h af[4];
#pragma unroll
    for (int mi = 0; mi < 4; ++mi) {
      const _Float16* ar = &As[waveM * 64 + mi * 16 + lr][0];
      af[mi] = cat8(*(const v8h*)(ar + hf * 8), *(const v8h*)(ar + 16 + hf * 8));
    }
#pragma unroll
    for (int ni = 0; ni < 2; ++ni) {
      const _Float16* br = &Bs[waveN * 32 + ni * 16 + lr][0];
      v16h bf = cat8(*(const v8h*)(br + hf * 8), *(const v8h*)(br + 16 + hf * 8));
#pragma unroll
      for (int mi = 0; mi < 4; ++mi)
        acc[mi][ni] = wmma_f16(af[mi], bf, acc[mi][ni]);
    }
    __syncthreads();
  }

  // ---- epilogue: bias + store ----
#pragma unroll
  for (int ni = 0; ni < 2; ++ni) {
    const int col = n0 + waveN * 32 + ni * 16 + lr;
    const float bv = bias[col];
#pragma unroll
    for (int mi = 0; mi < 4; ++mi) {
      const int rowb = m0 + waveM * 64 + mi * 16 + hf * 8;
#pragma unroll
      for (int r = 0; r < 8; ++r) {
        const float v = acc[mi][ni][r] + bv;
        if (OUT_IS_F16) ((_Float16*)Cv)[(size_t)(rowb + r) * N + col] = (_Float16)v;
        else            ((float*)Cv)[(size_t)(rowb + r) * N + col]    = v;
      }
    }
  }
}

// ---------------------------------------------------------------------------
// Flash attention: one block per (b, h, 64-query tile); 4 waves x 16 q-rows.
// K tile staged by the Tensor Data Mover (wave 0) while all waves transpose V.
// ---------------------------------------------------------------------------
__global__ void __launch_bounds__(128)
flash_attn_wmma(const _Float16* __restrict__ qkv, _Float16* __restrict__ attout)
{
  __shared__ _Float16 Ks[64][72];        // K tile row-major  [key][d] (TDM-padded)
  __shared__ _Float16 Vt[64][72];        // V tile transposed [d][key]
  __shared__ _Float16 Ps[4][16][72];     // per-wave P staging

  const int tid  = threadIdx.x;
  const int lane = tid & 31;
  const int wid  = tid >> 5;             // 0..3
  const int lr   = lane & 15;
  const int hf   = lane >> 4;

  const int qt = blockIdx.x & 31;
  const int bh = blockIdx.x >> 5;
  const int b  = bh >> 4;
  const int h  = bh & 15;

  // Q fragments (16x64) in registers; fold softmax scale 1/sqrt(64) into Q.
  const size_t rowQ = ((size_t)b * SEQ + qt * 64 + wid * 16 + lr) * (3 * EMBED);
  const int qcol = h * HDIM;
  v16h qf[2];
#pragma unroll
  for (int kk = 0; kk < 2; ++kk) {
    const _Float16* qp = qkv + rowQ + qcol + kk * 32;
    qf[kk] = cat8(*(const v8h*)(qp + hf * 8), *(const v8h*)(qp + 16 + hf * 8));
#pragma unroll
    for (int i = 0; i < 16; ++i) qf[kk][i] = qf[kk][i] * (_Float16)0.125f;
  }

  v8f o[4] = {};
  float m_i[8], l_i[8];
#pragma unroll
  for (int r = 0; r < 8; ++r) { m_i[r] = -1e30f; l_i[r] = 0.0f; }

  for (int kt = 0; kt < SEQ / 64; ++kt) {
    const size_t keyRow0 = (size_t)b * SEQ + kt * 64;

    // ---- K tile via TDM (wave 0 issues; DMA overlaps V staging below) ----
#if HAVE_TDM
    if (wid == 0) {
      tdm_load_tile64x64_f16((unsigned)(size_t)(&Ks[0][0]),
                             qkv + keyRow0 * (3 * EMBED) + EMBED + h * HDIM,
                             3 * EMBED, MROWS, 3 * EMBED);
    }
#else
    {
      const int key  = tid & 63;
      const int part = tid >> 6;
      const _Float16* kp0 = qkv + (keyRow0 + key) * (3 * EMBED) + EMBED + h * HDIM;
#pragma unroll
      for (int j = 0; j < 2; ++j) {
        const int off = part * 32 + j * 16;
        *(v8h*)(&Ks[key][off])     = *(const v8h*)(kp0 + off);
        *(v8h*)(&Ks[key][off + 8]) = *(const v8h*)(kp0 + off + 8);
      }
    }
#endif

    // ---- V tile transposed: key-pairs -> packed b32 LDS stores ----
    {
      const int kp = (tid & 31) * 2;       // key pair
      const int d0 = (tid >> 5) * 16;      // 16 d-values per thread
      const _Float16* vp = qkv + (keyRow0 + kp) * (3 * EMBED) + 2 * EMBED + h * HDIM + d0;
      v8h a0 = *(const v8h*)(vp);
      v8h a1 = *(const v8h*)(vp + 8);
      v8h b0 = *(const v8h*)(vp + 3 * EMBED);
      v8h b1 = *(const v8h*)(vp + 3 * EMBED + 8);
#pragma unroll
      for (int e = 0; e < 8; ++e) {
        v2h p; p[0] = a0[e]; p[1] = b0[e];
        *(v2h*)(&Vt[d0 + e][kp]) = p;
        v2h q; q[0] = a1[e]; q[1] = b1[e];
        *(v2h*)(&Vt[d0 + 8 + e][kp]) = q;
      }
    }
#if HAVE_TDM
    if (wid == 0) __builtin_amdgcn_s_wait_tensorcnt(0);
#endif
    __syncthreads();

    // ---- S = Q K^T ----
    v8f s[4] = {};
#pragma unroll
    for (int nt = 0; nt < 4; ++nt) {
#pragma unroll
      for (int kk = 0; kk < 2; ++kk) {
        const _Float16* kp = &Ks[nt * 16 + lr][kk * 32 + hf * 8];
        s[nt] = wmma_f16(qf[kk], cat8(*(const v8h*)kp, *(const v8h*)(kp + 16)), s[nt]);
      }
    }

    // ---- online softmax (row r lives in VGPR r across 16 lanes of a half) ----
#pragma unroll
    for (int r = 0; r < 8; ++r) {
      float mx = -1e30f;
#pragma unroll
      for (int nt = 0; nt < 4; ++nt) mx = fmaxf(mx, s[nt][r]);
#pragma unroll
      for (int off = 1; off < 16; off <<= 1) mx = fmaxf(mx, __shfl_xor(mx, off, 32));
      const float mn    = fmaxf(m_i[r], mx);
      const float alpha = __expf(m_i[r] - mn);
      m_i[r] = mn;
      float rs = 0.0f;
#pragma unroll
      for (int nt = 0; nt < 4; ++nt) {
        const float p = __expf(s[nt][r] - mn);
        s[nt][r] = p;
        rs += p;
      }
#pragma unroll
      for (int off = 1; off < 16; off <<= 1) rs += __shfl_xor(rs, off, 32);
      l_i[r] = l_i[r] * alpha + rs;
#pragma unroll
      for (int dt = 0; dt < 4; ++dt) o[dt][r] *= alpha;
    }

    // ---- P: C-layout -> per-wave LDS -> A-layout fragments ----
#pragma unroll
    for (int nt = 0; nt < 4; ++nt)
#pragma unroll
      for (int r = 0; r < 8; ++r)
        Ps[wid][hf * 8 + r][nt * 16 + lr] = (_Float16)s[nt][r];

    // ---- O += P V ----
#pragma unroll
    for (int kk = 0; kk < 2; ++kk) {
      const _Float16* pp = &Ps[wid][lr][kk * 32];
      v16h pf = cat8(*(const v8h*)(pp + hf * 8), *(const v8h*)(pp + 16 + hf * 8));
#pragma unroll
      for (int dt = 0; dt < 4; ++dt) {
        const _Float16* vp = &Vt[dt * 16 + lr][kk * 32 + hf * 8];
        o[dt] = wmma_f16(pf, cat8(*(const v8h*)vp, *(const v8h*)(vp + 16)), o[dt]);
      }
    }
    __syncthreads();
  }

  // ---- normalize + store to [B*S, EMBED] f16 ----
#pragma unroll
  for (int dt = 0; dt < 4; ++dt) {
#pragma unroll
    for (int r = 0; r < 8; ++r) {
      const int q = qt * 64 + wid * 16 + hf * 8 + r;
      attout[((size_t)b * SEQ + q) * EMBED + h * HDIM + dt * 16 + lr] =
          (_Float16)(o[dt][r] / l_i[r]);
    }
  }
}

// ---------------------------------------------------------------------------
extern "C" void kernel_launch(void* const* d_in, const int* in_sizes, int n_in,
                              void* d_out, int out_size, void* d_ws, size_t ws_size,
                              hipStream_t stream)
{
  (void)in_sizes; (void)n_in; (void)out_size;
  const float* x     = (const float*)d_in[0];
  const float* w_qkv = (const float*)d_in[1];
  const float* b_qkv = (const float*)d_in[2];
  const float* w_fc  = (const float*)d_in[3];
  const float* b_fc  = (const float*)d_in[4];
  float* out = (float*)d_out;

  _Float16* qkv = (_Float16*)d_ws;                      // [8192][3072] f16
  _Float16* att = qkv + (size_t)MROWS * (3 * EMBED);    // [8192][1024] f16
  if (ws_size < (size_t)MROWS * (3 * EMBED + EMBED) * sizeof(_Float16)) return;

  gemm128_wmma<true, true><<<dim3((3 * EMBED) / 128, MROWS / 128), dim3(256), 0, stream>>>(
      (const void*)x, w_qkv, b_qkv, (void*)qkv, MROWS, 3 * EMBED, EMBED);

  flash_attn_wmma<<<dim3(BATCH * HEADS * (SEQ / 64)), dim3(128), 0, stream>>>(qkv, att);

  gemm128_wmma<false, false><<<dim3(EMBED / 128, MROWS / 128), dim3(256), 0, stream>>>(
      (const void*)att, w_fc, b_fc, (void*)out, MROWS, EMBED, EMBED);
}
